// MultiHeadAttention_2284922601994
// MI455X (gfx1250) — matrix operand order repaired and checked by reading, not compiled, not run on hardware
//
#include <hip/hip_runtime.h>

// ---------------- CDNA5 WMMA types ----------------
typedef __attribute__((ext_vector_type(16))) __bf16       v16bf;
typedef __attribute__((ext_vector_type(8)))  float        v8f;
typedef __attribute__((ext_vector_type(4)))  unsigned int v4u;

typedef unsigned short u16;
typedef unsigned int   u32;

#define BB  4
#define NN  1024
#define MM  1024
#define DD  1024
#define HH  16
#define HDD 64

union Frag { v16bf v; v4u q[2]; };

__device__ __forceinline__ u16 f2bf(float f) {
  u32 u = __float_as_uint(f);
  u += 0x7FFFu + ((u >> 16) & 1u);   // round-to-nearest-even
  return (u16)(u >> 16);
}

#define WMMA_BF16(A, Bf, C) \
  __builtin_amdgcn_wmma_f32_16x16x32_bf16(false, (A).v, false, (Bf).v, (short)0, (C), false, false)

// 2x2 register-blocked 32x32 tile: rows {ar0, ar1(+16)}, cols {bc0, bc1(+16)}.
// A row-major (k contiguous); B pre-transposed so each column has k contiguous.
__device__ __forceinline__ void wmma_block2x2(const u16* __restrict__ ar0,
                                              const u16* __restrict__ ar1,
                                              const u16* __restrict__ bc0,
                                              const u16* __restrict__ bc1,
                                              int K, int kh,
                                              v8f& c00, v8f& c01, v8f& c10, v8f& c11) {
  for (int k0 = 0; k0 < K; k0 += 32) {
    Frag a0, a1, b0, b1;
    a0.q[0] = *(const v4u*)(ar0 + k0 + kh * 8);
    a0.q[1] = *(const v4u*)(ar0 + k0 + 16 + kh * 8);
    a1.q[0] = *(const v4u*)(ar1 + k0 + kh * 8);
    a1.q[1] = *(const v4u*)(ar1 + k0 + 16 + kh * 8);
    b0.q[0] = *(const v4u*)(bc0 + k0 + kh * 8);
    b0.q[1] = *(const v4u*)(bc0 + k0 + 16 + kh * 8);
    b1.q[0] = *(const v4u*)(bc1 + k0 + kh * 8);
    b1.q[1] = *(const v4u*)(bc1 + k0 + 16 + kh * 8);
    c00 = WMMA_BF16(a0, b0, c00);
    c01 = WMMA_BF16(a0, b1, c01);
    c10 = WMMA_BF16(a1, b0, c10);
    c11 = WMMA_BF16(a1, b1, c11);
  }
}

// ---------------- conversion / transpose ----------------
__global__ __launch_bounds__(256) void cvt_bf16_kernel(const float* __restrict__ src,
                                                       u16* __restrict__ dst, int n) {
  int i = blockIdx.x * blockDim.x + threadIdx.x;
  if (i < n) dst[i] = f2bf(src[i]);
}

// src is K x C row-major fp32; dst is C x K row-major bf16 (columns become rows)
__global__ __launch_bounds__(256) void transpose_bf16_kernel(const float* __restrict__ src,
                                                             u16* __restrict__ dst,
                                                             int K, int C) {
  int i = blockIdx.x * blockDim.x + threadIdx.x;
  if (i < K * C) {
    int k = i / C, c = i - k * C;
    dst[(size_t)c * K + k] = f2bf(src[i]);
  }
}

// ---------------- Q projection: (4096x1024) @ (1024x1024), 32x32 tiles ----------------
__global__ __launch_bounds__(256) void proj_q_kernel(const u16* __restrict__ X,
                                                     const u16* __restrict__ Wt,
                                                     const float* __restrict__ bias,
                                                     u16* __restrict__ Out) {
  int wave = (blockIdx.x * blockDim.x + threadIdx.x) >> 5;
  int lane = threadIdx.x & 31;
  int r = lane & 15, kh = lane >> 4;
  int rt = wave >> 5;        // 128 row tiles (32 rows each)
  int ct = wave & 31;        // 32  col tiles (32 cols each)
  int row0 = rt * 32, col0 = ct * 32;
  const u16* ar0 = X + (size_t)(row0 + r) * DD;
  const u16* bc0 = Wt + (size_t)(col0 + r) * DD;
  v8f c00 = {0.f,0.f,0.f,0.f,0.f,0.f,0.f,0.f}, c01 = c00, c10 = c00, c11 = c00;
  wmma_block2x2(ar0, ar0 + 16 * DD, bc0, bc0 + 16 * DD, DD, kh, c00, c01, c10, c11);
  int cA = col0 + r, cB = cA + 16;
  float bvA = bias[cA], bvB = bias[cB];
  int rbase = row0 + kh * 8;
#pragma unroll
  for (int j = 0; j < 8; ++j) {
    Out[(size_t)(rbase + j) * DD + cA]      = f2bf(c00[j] + bvA);
    Out[(size_t)(rbase + j) * DD + cB]      = f2bf(c01[j] + bvB);
    Out[(size_t)(rbase + 16 + j) * DD + cA] = f2bf(c10[j] + bvA);
    Out[(size_t)(rbase + 16 + j) * DD + cB] = f2bf(c11[j] + bvB);
  }
}

// ------- KV projection: (4096x1024) @ (1024x2048), 32x32 tiles, split K / V^T -------
__device__ __forceinline__ void kv_store(u16* __restrict__ Kout, u16* __restrict__ Vt,
                                         int rr, int c, float val) {
  if (c < DD) {                       // K: natural (b,m,h,hd) layout
    Kout[(size_t)rr * DD + c] = f2bf(val);
  } else {                            // V: transposed (b,h,hd,m) layout
    int c2 = c - DD, h = c2 >> 6, hd = c2 & 63;
    int b = rr >> 10, m = rr & 1023;
    Vt[((size_t)((b * HH + h) * HDD + hd)) * MM + m] = f2bf(val);
  }
}

__global__ __launch_bounds__(256) void proj_kv_kernel(const u16* __restrict__ X,
                                                      const u16* __restrict__ Wt,
                                                      const float* __restrict__ bias,
                                                      u16* __restrict__ Kout,
                                                      u16* __restrict__ Vt) {
  int wave = (blockIdx.x * blockDim.x + threadIdx.x) >> 5;
  int lane = threadIdx.x & 31;
  int r = lane & 15, kh = lane >> 4;
  int rt = wave >> 6;        // 128 row tiles
  int ct = wave & 63;        // 64  col tiles
  int row0 = rt * 32, col0 = ct * 32;
  const u16* ar0 = X + (size_t)(row0 + r) * DD;
  const u16* bc0 = Wt + (size_t)(col0 + r) * DD;
  v8f c00 = {0.f,0.f,0.f,0.f,0.f,0.f,0.f,0.f}, c01 = c00, c10 = c00, c11 = c00;
  wmma_block2x2(ar0, ar0 + 16 * DD, bc0, bc0 + 16 * DD, DD, kh, c00, c01, c10, c11);
  int cA = col0 + r, cB = cA + 16;
  float bvA = bias[cA], bvB = bias[cB];
  int rbase = row0 + kh * 8;
#pragma unroll
  for (int j = 0; j < 8; ++j) {
    kv_store(Kout, Vt, rbase + j,      cA, c00[j] + bvA);
    kv_store(Kout, Vt, rbase + j,      cB, c01[j] + bvB);
    kv_store(Kout, Vt, rbase + 16 + j, cA, c10[j] + bvA);
    kv_store(Kout, Vt, rbase + 16 + j, cB, c11[j] + bvB);
  }
}

// ------- scores = scale * Q K^T, masked, 32(n)x16(m) per step, written 'bnmh' -------
__global__ __launch_bounds__(256) void qk_scores_kernel(const u16* __restrict__ Qb,
                                                        const u16* __restrict__ Kb,
                                                        const unsigned char* __restrict__ mask,
                                                        float* __restrict__ attn) {
  int wave = (blockIdx.x * blockDim.x + threadIdx.x) >> 5;
  int lane = threadIdx.x & 31;
  int r = lane & 15, kh = lane >> 4;
  int nt = wave & 31;               // 32 row-blocks of 32 rows
  int h  = (wave >> 5) & 15;
  int b  = wave >> 9;
  int n0 = nt * 32;
  const u16* ar0 = Qb + (size_t)(b * NN + n0 + r) * DD + h * HDD;
  const u16* ar1 = ar0 + 16 * DD;
  Frag a00, a01, a10, a11;          // hoist both k-steps (HD=64) for both row blocks
  a00.q[0] = *(const v4u*)(ar0 + kh * 8);
  a00.q[1] = *(const v4u*)(ar0 + 16 + kh * 8);
  a01.q[0] = *(const v4u*)(ar0 + 32 + kh * 8);
  a01.q[1] = *(const v4u*)(ar0 + 48 + kh * 8);
  a10.q[0] = *(const v4u*)(ar1 + kh * 8);
  a10.q[1] = *(const v4u*)(ar1 + 16 + kh * 8);
  a11.q[0] = *(const v4u*)(ar1 + 32 + kh * 8);
  a11.q[1] = *(const v4u*)(ar1 + 48 + kh * 8);
  const float scale = 0.125f;       // HD^-0.5 = 1/8
  size_t abase = (size_t)b * NN * MM * HH;
  for (int mt = 0; mt < MM / 16; ++mt) {
    int m0 = mt * 16;
    const u16* bcol = Kb + (size_t)(b * MM + m0 + r) * DD + h * HDD;
    Frag b0, b1;
    b0.q[0] = *(const v4u*)(bcol + kh * 8);
    b0.q[1] = *(const v4u*)(bcol + 16 + kh * 8);
    b1.q[0] = *(const v4u*)(bcol + 32 + kh * 8);
    b1.q[1] = *(const v4u*)(bcol + 48 + kh * 8);
    v8f acc0 = {0.f,0.f,0.f,0.f,0.f,0.f,0.f,0.f}, acc1 = acc0;
    acc0 = WMMA_BF16(a00, b0, acc0);
    acc0 = WMMA_BF16(a01, b1, acc0);
    acc1 = WMMA_BF16(a10, b0, acc1);
    acc1 = WMMA_BF16(a11, b1, acc1);
    int m = m0 + r;
    bool msk = mask[b * MM + m] != 0;
#pragma unroll
    for (int j = 0; j < 8; ++j) {
      int n = n0 + kh * 8 + j;
      float v0 = msk ? -__builtin_inff() : acc0[j] * scale;
      float v1 = msk ? -__builtin_inff() : acc1[j] * scale;
      attn[abase + ((size_t)n * MM + m) * HH + h]        = v0;
      attn[abase + ((size_t)(n + 16) * MM + m) * HH + h] = v1;
    }
  }
}

// ------- softmax over m for each (b,n,h); also emit bf16 probs (b,h,n,m) -------
__global__ __launch_bounds__(256) void softmax_kernel(float* __restrict__ attn,
                                                      u16* __restrict__ Pb) {
  int bn = blockIdx.x;
  int b = bn >> 10, n = bn & 1023;
  int tid = threadIdx.x;
  int h = tid & 15, g = tid >> 4;           // 16 groups x 16 heads
  float* base = attn + (size_t)bn * (MM * HH);   // contiguous M x H slice
  float vals[64];
  float mx = -__builtin_inff();
#pragma unroll
  for (int j = 0; j < 64; ++j) {
    int m = g + j * 16;
    float v = base[(size_t)m * HH + h];
    vals[j] = v;
    mx = fmaxf(mx, v);
  }
  __shared__ float red[16][17];
  red[g][h] = mx;
  __syncthreads();
  if (g == 0) {
    float t = red[0][h];
#pragma unroll
    for (int i = 1; i < 16; ++i) t = fmaxf(t, red[i][h]);
    red[0][h] = t;
  }
  __syncthreads();
  mx = red[0][h];
  __syncthreads();
  float s = 0.f;
#pragma unroll
  for (int j = 0; j < 64; ++j) {
    float e = __expf(vals[j] - mx);
    vals[j] = e;
    s += e;
  }
  red[g][h] = s;
  __syncthreads();
  if (g == 0) {
    float t = 0.f;
#pragma unroll
    for (int i = 0; i < 16; ++i) t += red[i][h];
    red[0][h] = t;
  }
  __syncthreads();
  float inv = 1.0f / red[0][h];
  u16* pb = Pb + ((size_t)(b * HH + h) * NN + n) * MM;
#pragma unroll
  for (int j = 0; j < 64; ++j) {
    int m = g + j * 16;
    float p = vals[j] * inv;
    base[(size_t)m * HH + h] = p;
    pb[m] = f2bf(p);
  }
}

// ------- AV: per (b,h): (N x M) @ (M x HD), 32x32 tiles -> O (b,n,h,hd) fp32 -------
__global__ __launch_bounds__(256) void av_kernel(const u16* __restrict__ Pb,
                                                 const u16* __restrict__ Vt,
                                                 float* __restrict__ Ob) {
  int wave = (blockIdx.x * blockDim.x + threadIdx.x) >> 5;
  int lane = threadIdx.x & 31;
  int r = lane & 15, kh = lane >> 4;
  int ct = wave & 1;                // 2 col tiles of 32 (HD=64)
  int nt = (wave >> 1) & 31;        // 32 row tiles of 32
  int h  = (wave >> 6) & 15;
  int b  = wave >> 10;
  int n0 = nt * 32, c0 = ct * 32;
  const u16* ar0 = Pb + ((size_t)((b * HH + h) * NN) + n0 + r) * MM;
  const u16* bc0 = Vt + ((size_t)((b * HH + h) * HDD) + c0 + r) * MM;
  v8f c00 = {0.f,0.f,0.f,0.f,0.f,0.f,0.f,0.f}, c01 = c00, c10 = c00, c11 = c00;
  wmma_block2x2(ar0, ar0 + 16 * MM, bc0, bc0 + 16 * MM, MM, kh, c00, c01, c10, c11);
  int hdA = c0 + r, hdB = hdA + 16;
  int rbase = n0 + kh * 8;
#pragma unroll
  for (int j = 0; j < 8; ++j) {
    Ob[(size_t)(b * NN + rbase + j) * DD + h * HDD + hdA]      = c00[j];
    Ob[(size_t)(b * NN + rbase + j) * DD + h * HDD + hdB]      = c01[j];
    Ob[(size_t)(b * NN + rbase + 16 + j) * DD + h * HDD + hdA] = c10[j];
    Ob[(size_t)(b * NN + rbase + 16 + j) * DD + h * HDD + hdB] = c11[j];
  }
}

// ------------- final projection: (4096x1024) @ (1024x2) + bp -------------
__global__ __launch_bounds__(256) void final_proj_kernel(const float* __restrict__ Ob,
                                                         const float* __restrict__ Wp,
                                                         const float* __restrict__ bp,
                                                         float* __restrict__ out) {
  int wave = (blockIdx.x * blockDim.x + threadIdx.x) >> 5;
  int lane = threadIdx.x & 31;
  const float* row = Ob + (size_t)wave * DD;
  float a0 = 0.f, a1 = 0.f;
  for (int d = lane; d < DD; d += 32) {
    float o = row[d];
    a0 += o * Wp[2 * d];
    a1 += o * Wp[2 * d + 1];
  }
#pragma unroll
  for (int off = 16; off > 0; off >>= 1) {
    a0 += __shfl_down(a0, off, 32);
    a1 += __shfl_down(a1, off, 32);
  }
  if (lane == 0) {
    out[wave * 2]     = a0 + bp[0];
    out[wave * 2 + 1] = a1 + bp[1];
  }
}

extern "C" void kernel_launch(void* const* d_in, const int* in_sizes, int n_in,
                              void* d_out, int out_size, void* d_ws, size_t ws_size,
                              hipStream_t stream) {
  const float* query          = (const float*)d_in[0];
  const float* key_value      = (const float*)d_in[1];
  const unsigned char* mask   = (const unsigned char*)d_in[2];  // jax bool_ = 1 byte
  const float* Wq             = (const float*)d_in[3];
  const float* bq             = (const float*)d_in[4];
  const float* Wkv            = (const float*)d_in[5];
  const float* bkv            = (const float*)d_in[6];
  const float* Wp             = (const float*)d_in[7];
  const float* bp             = (const float*)d_in[8];

  float* out  = (float*)d_out;
  float* attn = out + (size_t)BB * NN * 2;   // outputs first, then attention

  // workspace layout (bf16 unless noted), ~190 MB total
  u16*   Xq   = (u16*)d_ws;                              // 4M elems
  u16*   Xkv  = Xq   + (size_t)BB * NN * DD;             // 4M
  u16*   WqT  = Xkv  + (size_t)BB * MM * DD;             // 1M
  u16*   WkvT = WqT  + (size_t)DD * DD;                  // 2M
  u16*   Qb   = WkvT + (size_t)DD * 2 * DD;              // 4M
  u16*   Kb   = Qb   + (size_t)BB * NN * DD;             // 4M
  u16*   Vt   = Kb   + (size_t)BB * MM * DD;             // 4M
  u16*   Pb   = Vt   + (size_t)BB * HH * HDD * MM;       // 64M
  float* Ob   = (float*)(Pb + (size_t)BB * HH * NN * MM);// 4M fp32

  int nq = BB * NN * DD;  // 4194304
  cvt_bf16_kernel<<<(nq + 255) / 256, 256, 0, stream>>>(query, Xq, nq);
  cvt_bf16_kernel<<<(nq + 255) / 256, 256, 0, stream>>>(key_value, Xkv, nq);
  transpose_bf16_kernel<<<(DD * DD + 255) / 256, 256, 0, stream>>>(Wq, WqT, DD, DD);
  transpose_bf16_kernel<<<(DD * 2 * DD + 255) / 256, 256, 0, stream>>>(Wkv, WkvT, DD, 2 * DD);

  proj_q_kernel<<<512, 256, 0, stream>>>(Xq, WqT, bq, Qb);           // 4096 32x32 tiles
  proj_kv_kernel<<<1024, 256, 0, stream>>>(Xkv, WkvT, bkv, Kb, Vt);  // 8192 32x32 tiles
  qk_scores_kernel<<<256, 256, 0, stream>>>(Qb, Kb, mask, attn);     // 2048 waves x 64 m-tiles
  softmax_kernel<<<4096, 256, 0, stream>>>(attn, Pb);                // one block per (b,n)
  av_kernel<<<512, 256, 0, stream>>>(Pb, Vt, Ob);                    // 4096 32x32 tiles
  final_proj_kernel<<<512, 256, 0, stream>>>(Ob, Wp, bp, out);       // one wave per row
}
